// Classifiers_18133351923788
// MI455X (gfx1250) — matrix operand order repaired and checked by reading, not compiled, not run on hardware
//
#include <hip/hip_runtime.h>
#include <hip/hip_bf16.h>

typedef __attribute__((ext_vector_type(16))) _Float16 v16h;
typedef __attribute__((ext_vector_type(8)))  _Float16 v8h;
typedef __attribute__((ext_vector_type(8)))  float    v8f;
typedef __attribute__((ext_vector_type(4)))  float    v4f;
typedef __attribute__((ext_vector_type(4)))  int      v4i;

#define TILE_M 128
#define TILE_N 64
#define TILE_K 32
#define LDA_P  40   // 32 + 8 halves pad; 80B row stride (16B multiple)
#define LDB_P  40   // 32 + 8 halves pad (B stored transposed [N][K])

#if __has_builtin(__builtin_amdgcn_global_load_async_to_lds_b128) && \
    __has_builtin(__builtin_amdgcn_s_wait_asynccnt)
#define USE_ASYNC 1
#else
#define USE_ASYNC 0
#endif

typedef __attribute__((address_space(1))) v4i gv4i;  // global 16B chunk (non-const per builtin sig)
typedef __attribute__((address_space(3))) v4i lv4i;  // LDS 16B chunk

union U16 { v16h v; v8h h[2]; };

// ---------------------------------------------------------------------------
// WMMA GEMM: Out[M,N] = op( A[M,K] * B[K,N] + bias + addsrc )
//   B is supplied TRANSPOSED: BwT[N][K] with row stride ldb (= Kpad).
// REQUIRES: M % 128 == 0, N % 64 == 0, K % 32 == 0 (all call sites padded).
// Block = 256 threads = 8 wave32; each wave computes a 16x64 strip (4 WMMA/K-step).
// ---------------------------------------------------------------------------
template<typename AT, bool RELU, bool OUT16, bool ADD>
__global__ __launch_bounds__(256)
void gemm_wmma_kernel(const AT* __restrict__ A, long lda,
                      const _Float16* __restrict__ BwT, long ldb,
                      const float* __restrict__ bias,
                      const float* __restrict__ addsrc, long ldadd,
                      void* __restrict__ Out, long ldout,
                      int M, int N, int K)
{
    __shared__ alignas(16) _Float16 lsA[TILE_M][LDA_P];   // [m][k]
    __shared__ alignas(16) _Float16 lsB[TILE_N][LDB_P];   // [n][k] (transposed tile)

    const int tid  = threadIdx.x;
    const int lane = tid & 31;
    const int wv   = tid >> 5;                 // wave id 0..7
    const int m0   = blockIdx.y * TILE_M;
    const int n0   = blockIdx.x * TILE_N;
    const int mlo  = lane & 15;
    const int hi   = lane >> 4;                // 0 or 1

    // staging coordinates (per thread, loop-invariant)
    const int arow = tid >> 1;                 // A: 128 rows x 32 halves, 16/thread
    const int acol = (tid & 1) * 16;
    const int brow = tid >> 2;                 // B: 64 rows x 32 halves, 8/thread
    const int bcol = (tid & 3) * 8;

    v8f c[4] = {v8f{0}, v8f{0}, v8f{0}, v8f{0}};

    for (int k0 = 0; k0 < K; k0 += TILE_K) {
        // ---- stage A tile (vectorized, no guards)
        if constexpr (sizeof(AT) == 2) {
            const _Float16* ga = (const _Float16*)(const void*)A
                               + (long)(m0 + arow) * lda + (k0 + acol);
#if USE_ASYNC
            __builtin_amdgcn_global_load_async_to_lds_b128(
                (gv4i*)ga,       (lv4i*)&lsA[arow][acol],     0, 0);
            __builtin_amdgcn_global_load_async_to_lds_b128(
                (gv4i*)(ga + 8), (lv4i*)&lsA[arow][acol + 8], 0, 0);
#else
            *(v8h*)&lsA[arow][acol]     = *(const v8h*)ga;
            *(v8h*)&lsA[arow][acol + 8] = *(const v8h*)(ga + 8);
#endif
        } else {
            const float* ga = (const float*)(const void*)A
                            + (long)(m0 + arow) * lda + (k0 + acol);
            v4f f0 = *(const v4f*)(ga + 0);
            v4f f1 = *(const v4f*)(ga + 4);
            v4f f2 = *(const v4f*)(ga + 8);
            v4f f3 = *(const v4f*)(ga + 12);
            v8h h0, h1;
            #pragma unroll
            for (int i = 0; i < 4; ++i) {
                h0[i]     = (_Float16)f0[i];
                h0[4 + i] = (_Float16)f1[i];
                h1[i]     = (_Float16)f2[i];
                h1[4 + i] = (_Float16)f3[i];
            }
            *(v8h*)&lsA[arow][acol]     = h0;
            *(v8h*)&lsA[arow][acol + 8] = h1;
        }
        // ---- stage B tile from transposed weights (vectorized, no guards)
        {
            const _Float16* gb = BwT + (long)(n0 + brow) * ldb + (k0 + bcol);
#if USE_ASYNC
            __builtin_amdgcn_global_load_async_to_lds_b128(
                (gv4i*)gb, (lv4i*)&lsB[brow][bcol], 0, 0);
#else
            *(v8h*)&lsB[brow][bcol] = *(const v8h*)gb;
#endif
        }
#if USE_ASYNC
        __builtin_amdgcn_s_wait_asynccnt(0);
#endif
        __syncthreads();

        // ---- A fragment: lane<16 -> K {0..7,16..23}; lane>=16 -> K {8..15,24..31}
        U16 au;
        {
            const int r  = wv * 16 + mlo;
            const int ka = hi * 8;
            au.h[0] = *(const v8h*)&lsA[r][ka];
            au.h[1] = *(const v8h*)&lsA[r][16 + ka];
        }
        // ---- 4 N-subtiles; B fragment: lane<16 -> K {0..7,16..23}; lane>=16 -> K {8..15,24..31}
        #pragma unroll
        for (int j = 0; j < 4; ++j) {
            U16 bu;
            const int n  = j * 16 + mlo;
            const int kb = hi * 8;
            bu.h[0] = *(const v8h*)&lsB[n][kb];
            bu.h[1] = *(const v8h*)&lsB[n][16 + kb];
            c[j] = __builtin_amdgcn_wmma_f32_16x16x32_f16(
                false, au.v, false, bu.v, (short)0, c[j], false, false);
        }
        __syncthreads();
    }

    // ---- epilogue: C layout: v_r -> M = r + 8*hi, N = lane%16
    #pragma unroll
    for (int j = 0; j < 4; ++j) {
        const int gn = n0 + j * 16 + mlo;
        const float bb = bias ? bias[gn] : 0.f;
        #pragma unroll
        for (int r = 0; r < 8; ++r) {
            const long gm = m0 + wv * 16 + hi * 8 + r;
            float v = c[j][r] + bb;
            if (ADD)  v += addsrc[gm * ldadd + gn];
            if (RELU) v = v > 0.f ? v : 0.f;
            if (OUT16) ((_Float16*)Out)[gm * ldout + gn] = (_Float16)v;
            else       ((float*)Out)[gm * ldout + gn]    = v;
        }
    }
}

// ---------------------------------------------------------------------------
// LSTM gate nonlinearities + state update (keras order i,f,g,o; act=relu)
// ---------------------------------------------------------------------------
__global__ void lstm_pointwise(const float* __restrict__ z, float* __restrict__ cst,
                               _Float16* __restrict__ h, _Float16* __restrict__ hs,
                               int B, int U, int T, int t)
{
    int idx = blockIdx.x * blockDim.x + threadIdx.x;
    if (idx >= B * U) return;
    int b = idx / U, k = idx % U;
    const float* zr = z + (long)b * 4 * U;
    float ig = 1.f / (1.f + __expf(-zr[k]));
    float fg = 1.f / (1.f + __expf(-zr[U + k]));
    float g  = zr[2 * U + k]; g = g > 0.f ? g : 0.f;
    float og = 1.f / (1.f + __expf(-zr[3 * U + k]));
    float cn = fg * cst[idx] + ig * g;
    cst[idx] = cn;
    float rc = cn > 0.f ? cn : 0.f;
    float hv = og * rc;
    h[idx] = (_Float16)hv;
    if (hs) hs[((long)b * T + t) * U + k] = (_Float16)hv;
}

__global__ void zero_state(_Float16* h, float* c, int n)
{
    int i = blockIdx.x * blockDim.x + threadIdx.x;
    if (i < n) { h[i] = (_Float16)0.f; c[i] = 0.f; }
}

// f32 [K,N] -> TRANSPOSED f16 [Npad][Kpad] with zero padding
__global__ void cvt_pad_f16_T(const float* __restrict__ src, _Float16* __restrict__ dst,
                              int K, int N, int Kpad, int Npad)
{
    int i = blockIdx.x * blockDim.x + threadIdx.x;
    if (i >= Npad * Kpad) return;
    int n = i / Kpad, k = i % Kpad;
    float v = (k < K && n < N) ? src[k * N + n] : 0.f;
    dst[i] = (_Float16)v;
}

__global__ void pad_bias(const float* __restrict__ src, float* __restrict__ dst, int N, int Npad)
{
    int i = blockIdx.x * blockDim.x + threadIdx.x;
    if (i >= Npad) return;
    dst[i] = (i < N) ? src[i] : 0.f;
}

// BN folded into lstm1 input weights, stored transposed: dst[n][d] = scale(d)*W[d,n]
__global__ void bn_fold_w_T(const float* __restrict__ W, const float* __restrict__ g,
                            const float* __restrict__ va,
                            _Float16* __restrict__ dst, int D, int Dpad, int N)
{
    int i = blockIdx.x * blockDim.x + threadIdx.x;
    if (i >= N * Dpad) return;
    int n = i / Dpad, d = i % Dpad;
    float v = 0.f;
    if (d < D) { float s = g[d] * rsqrtf(va[d] + 1e-3f); v = s * W[d * N + n]; }
    dst[i] = (_Float16)v;
}

// b'[n] = b[n] + sum_d (beta - mean*scale)[d] * W[d,n]
__global__ void bn_fold_b(const float* __restrict__ W, const float* __restrict__ b,
                          const float* __restrict__ g, const float* __restrict__ be,
                          const float* __restrict__ mu, const float* __restrict__ va,
                          float* __restrict__ dst, int D, int N)
{
    int n = blockIdx.x * blockDim.x + threadIdx.x;
    if (n >= N) return;
    float acc = b[n];
    for (int d = 0; d < D; ++d) {
        float s = g[d] * rsqrtf(va[d] + 1e-3f);
        acc += (be[d] - mu[d] * s) * W[d * N + n];
    }
    dst[n] = acc;
}

// d2 + softmax: one thread per row (C <= 11, Kd = 100)
__global__ void head_softmax(const _Float16* __restrict__ A1, int lda,
                             const float* __restrict__ W, const float* __restrict__ b,
                             float* __restrict__ out, int B, int Kd, int C)
{
    int r = blockIdx.x * blockDim.x + threadIdx.x;
    if (r >= B) return;
    float logits[16];
    for (int c = 0; c < C; ++c) logits[c] = b[c];
    for (int k = 0; k < Kd; ++k) {
        float a = (float)A1[(long)r * lda + k];
        for (int c = 0; c < C; ++c) logits[c] += a * W[k * C + c];
    }
    float mx = logits[0];
    for (int c = 1; c < C; ++c) mx = fmaxf(mx, logits[c]);
    float s = 0.f;
    for (int c = 0; c < C; ++c) { float e = __expf(logits[c] - mx); logits[c] = e; s += e; }
    float inv = 1.f / s;
    for (int c = 0; c < C; ++c) out[(long)r * C + c] = logits[c] * inv;
}

// ---------------------------------------------------------------------------
// Input index map (setup_inputs dict order, params flattened depth-first)
// ---------------------------------------------------------------------------
enum {
    IN_AC = 0, IN_CC = 1,
    D0_W = 2, D0_B = 3,
    BNA_G = 4, BNA_B = 5, BNA_M = 6, BNA_V = 7,
    BNC_G = 8, BNC_B = 9, BNC_M = 10, BNC_V = 11,
    L1A_W = 12, L1A_U = 13, L1A_B = 14,
    L2A_W = 15, L2A_U = 16, L2A_B = 17,
    L3A_W = 18, L3A_U = 19, L3A_B = 20,
    D1A_W = 21, D1A_B = 22, D2A_W = 23, D2A_B = 24,
    L1C_W = 25, L1C_U = 26, L1C_B = 27,
    L2C_W = 28, L2C_U = 29, L2C_B = 30,
    L3C_W = 31, L3C_U = 32, L3C_B = 33,
    D1C_W = 34, D1C_B = 35, D2C_W = 36, D2C_B = 37
};

static inline dim3 gemm_grid(int M, int N)
{
    return dim3((N + TILE_N - 1) / TILE_N, (M + TILE_M - 1) / TILE_M);
}

extern "C" void kernel_launch(void* const* d_in, const int* in_sizes, int n_in,
                              void* d_out, int out_size, void* d_ws, size_t ws_size,
                              hipStream_t stream)
{
    (void)in_sizes; (void)n_in; (void)out_size; (void)ws_size;
    const int B = 8192, Tmax = 5, BTmax = B * Tmax;

    const float* input_ac = (const float*)d_in[IN_AC];
    const float* input_cc = (const float*)d_in[IN_CC];

    // ---- workspace carve (bump allocator, 256B aligned) ----
    char* p = (char*)d_ws;
    auto alloc = [&](size_t bytes) -> void* {
        void* r = (void*)p;
        p += (bytes + 255) & ~(size_t)255;
        return r;
    };
    _Float16* wY0  = (_Float16*)alloc((size_t)BTmax * 320 * 2); // dense0 out (300->320)
    float*    wZx  = (float*)   alloc((size_t)BTmax * 512 * 4); // input projections (reused)
    _Float16* wHsA = (_Float16*)alloc((size_t)BTmax * 64 * 2);  // lstm1 hidden seq
    _Float16* wHsB = (_Float16*)alloc((size_t)BTmax * 128 * 2); // lstm2 hidden seq
    float*    wZ   = (float*)   alloc((size_t)B * 512 * 4);     // per-step gate preacts
    _Float16* wH   = (_Float16*)alloc((size_t)B * 128 * 2);     // h state
    float*    wC   = (float*)   alloc((size_t)B * 128 * 4);     // c state
    _Float16* wA1  = (_Float16*)alloc((size_t)B * 128 * 2);     // d1 out (100->128)
    // transposed f16 weights [Npad][Kpad]
    _Float16* wW0  = (_Float16*)alloc((size_t)320 * 512 * 2);
    float*    wb0  = (float*)   alloc(320 * 4);
    _Float16* wW1p = (_Float16*)alloc((size_t)256 * 320 * 2);   // BN-folded lstm1 W
    float*    wb1p = (float*)   alloc(256 * 4);
    _Float16* wU1  = (_Float16*)alloc((size_t)256 * 64 * 2);
    _Float16* wW2  = (_Float16*)alloc((size_t)512 * 64 * 2);
    _Float16* wU2  = (_Float16*)alloc((size_t)512 * 128 * 2);
    _Float16* wW3  = (_Float16*)alloc((size_t)256 * 128 * 2);
    _Float16* wU3  = (_Float16*)alloc((size_t)256 * 64 * 2);
    _Float16* wW4  = (_Float16*)alloc((size_t)128 * 64 * 2);
    float*    wb4  = (float*)   alloc(128 * 4);

    auto cvtT = [&](const float* src, _Float16* dst, int K, int N, int Kp, int Np) {
        int tot = Np * Kp;
        cvt_pad_f16_T<<<(tot + 255) / 256, 256, 0, stream>>>(src, dst, K, N, Kp, Np);
    };

    // ---- shared dense0 weight prep (once) ----
    cvtT((const float*)d_in[D0_W], wW0, 512, 300, 512, 320);
    pad_bias<<<2, 256, 0, stream>>>((const float*)d_in[D0_B], wb0, 300, 320);

    // ---- one branch ----
    auto run_branch = [&](const float* input, int T,
                          const float* bng, const float* bnb, const float* bnm, const float* bnv,
                          const float* W1, const float* U1, const float* b1,
                          const float* W2, const float* U2, const float* b2,
                          const float* W3, const float* U3, const float* b3,
                          const float* W4, const float* b4,
                          const float* W5, const float* b5,
                          int C, float* outp) {
        const int BT = B * T;

        // per-branch weight prep (BN folded into lstm1 input projection)
        bn_fold_w_T<<<(256 * 320 + 255) / 256, 256, 0, stream>>>(W1, bng, bnv, wW1p, 300, 320, 256);
        bn_fold_b<<<(256 + 63) / 64, 64, 0, stream>>>(W1, b1, bng, bnb, bnm, bnv, wb1p, 300, 256);
        cvtT(U1, wU1, 64, 256, 64, 256);
        cvtT(W2, wW2, 64, 512, 64, 512);
        cvtT(U2, wU2, 128, 512, 128, 512);
        cvtT(W3, wW3, 128, 256, 128, 256);
        cvtT(U3, wU3, 64, 256, 64, 256);
        cvtT(W4, wW4, 64, 100, 64, 128);
        pad_bias<<<1, 128, 0, stream>>>(b4, wb4, 100, 128);

        // dense0 + relu  (BN applied via folded lstm1 weights)
        gemm_wmma_kernel<float, true, true, false>
            <<<gemm_grid(BT, 320), 256, 0, stream>>>(
                input, 512, wW0, 512, wb0, nullptr, 0, wY0, 320, BT, 320, 512);

        // ---- LSTM1: 300(320)->64 ----
        gemm_wmma_kernel<_Float16, false, false, false>
            <<<gemm_grid(BT, 256), 256, 0, stream>>>(
                wY0, 320, wW1p, 320, wb1p, nullptr, 0, wZx, 256, BT, 256, 320);
        zero_state<<<(B * 64 + 255) / 256, 256, 0, stream>>>(wH, wC, B * 64);
        for (int t = 0; t < T; ++t) {
            gemm_wmma_kernel<_Float16, false, false, true>
                <<<gemm_grid(B, 256), 256, 0, stream>>>(
                    wH, 64, wU1, 64, nullptr, wZx + (long)t * 256, (long)T * 256,
                    wZ, 256, B, 256, 64);
            lstm_pointwise<<<(B * 64 + 255) / 256, 256, 0, stream>>>(
                wZ, wC, wH, wHsA, B, 64, T, t);
        }

        // ---- LSTM2: 64->128 ----
        gemm_wmma_kernel<_Float16, false, false, false>
            <<<gemm_grid(BT, 512), 256, 0, stream>>>(
                wHsA, 64, wW2, 64, b2, nullptr, 0, wZx, 512, BT, 512, 64);
        zero_state<<<(B * 128 + 255) / 256, 256, 0, stream>>>(wH, wC, B * 128);
        for (int t = 0; t < T; ++t) {
            gemm_wmma_kernel<_Float16, false, false, true>
                <<<gemm_grid(B, 512), 256, 0, stream>>>(
                    wH, 128, wU2, 128, nullptr, wZx + (long)t * 512, (long)T * 512,
                    wZ, 512, B, 512, 128);
            lstm_pointwise<<<(B * 128 + 255) / 256, 256, 0, stream>>>(
                wZ, wC, wH, wHsB, B, 128, T, t);
        }

        // ---- LSTM3: 128->64 (final h only) ----
        gemm_wmma_kernel<_Float16, false, false, false>
            <<<gemm_grid(BT, 256), 256, 0, stream>>>(
                wHsB, 128, wW3, 128, b3, nullptr, 0, wZx, 256, BT, 256, 128);
        zero_state<<<(B * 64 + 255) / 256, 256, 0, stream>>>(wH, wC, B * 64);
        for (int t = 0; t < T; ++t) {
            gemm_wmma_kernel<_Float16, false, false, true>
                <<<gemm_grid(B, 256), 256, 0, stream>>>(
                    wH, 64, wU3, 64, nullptr, wZx + (long)t * 256, (long)T * 256,
                    wZ, 256, B, 256, 64);
            lstm_pointwise<<<(B * 64 + 255) / 256, 256, 0, stream>>>(
                wZ, wC, wH, nullptr, B, 64, T, t);
        }

        // ---- d1 (relu), N padded to 128 ----
        gemm_wmma_kernel<_Float16, true, true, false>
            <<<gemm_grid(B, 128), 256, 0, stream>>>(
                wH, 64, wW4, 64, wb4, nullptr, 0, wA1, 128, B, 128, 64);

        // ---- d2 + softmax ----
        head_softmax<<<(B + 127) / 128, 128, 0, stream>>>(
            wA1, 128, W5, b5, outp, B, 100, C);
    };

    float* out_ac = (float*)d_out;
    float* out_cc = (float*)d_out + (size_t)B * 11;

    run_branch(input_ac, 3,
               (const float*)d_in[BNA_G], (const float*)d_in[BNA_B],
               (const float*)d_in[BNA_M], (const float*)d_in[BNA_V],
               (const float*)d_in[L1A_W], (const float*)d_in[L1A_U], (const float*)d_in[L1A_B],
               (const float*)d_in[L2A_W], (const float*)d_in[L2A_U], (const float*)d_in[L2A_B],
               (const float*)d_in[L3A_W], (const float*)d_in[L3A_U], (const float*)d_in[L3A_B],
               (const float*)d_in[D1A_W], (const float*)d_in[D1A_B],
               (const float*)d_in[D2A_W], (const float*)d_in[D2A_B],
               11, out_ac);

    run_branch(input_cc, 5,
               (const float*)d_in[BNC_G], (const float*)d_in[BNC_B],
               (const float*)d_in[BNC_M], (const float*)d_in[BNC_V],
               (const float*)d_in[L1C_W], (const float*)d_in[L1C_U], (const float*)d_in[L1C_B],
               (const float*)d_in[L2C_W], (const float*)d_in[L2C_U], (const float*)d_in[L2C_B],
               (const float*)d_in[L3C_W], (const float*)d_in[L3C_U], (const float*)d_in[L3C_B],
               (const float*)d_in[D1C_W], (const float*)d_in[D1C_B],
               (const float*)d_in[D2C_W], (const float*)d_in[D2C_B],
               10, out_cc);
}
